// DLRM_44427141710336
// MI455X (gfx1250) — matrix operand order repaired and checked by reading, not compiled, not run on hardware
//
#include <hip/hip_runtime.h>

typedef _Float16 half_t;
typedef __attribute__((ext_vector_type(16))) half_t v16h;
typedef __attribute__((ext_vector_type(8)))  float  v8f;

#define B_DIM 4096
#define C_DIM 100
#define NROWS (B_DIM * C_DIM)      // 409600 = 32 * 12800
#define NSUPER (NROWS / 32)        // 12800 32-row super-tiles
#define TOP_BLOCKS 400
#define TOP_WAVES 8                // waves per block (2 per SIMD32)

// swizzled-weight pool (halves): W0 | W1 | W2 | Wb1
#define OFF_W0   0                 // 32 frags * 512
#define OFF_W1   16384             // 64 frags * 512
#define OFF_W2   49152             // 16 frags * 512
#define OFF_WB1  57344             // 2  frags * 512
#define W_HALVES 58368             // 116,736 bytes total

// per-wave slab (halves): xh/h3 [32][64] @0 ; h1 [32][256] @2048
// (hbuf [32][32] and h2 [32][128] alias h1) ; ivec f32 [32][32] @10240
#define SLAB_HALVES 12288          // 24,576 B; total LDS = 313,344 B <= 320 KB

// ---------------------------------------------------------------- WMMA helpers

__device__ __forceinline__ v8f wmma16(v16h a, v16h b, v8f c) {
    return __builtin_amdgcn_wmma_f32_16x16x32_f16(false, a, false, b, (short)0, c,
                                                  false, false);
}

// A fragment (16x32 f16, row-major, leading dim ld) per ISA 16-bit A layout:
// lanes 0-15: M=lane, VGPR0-3: K=0..7, VGPR4-7: K=16..23; lanes 16-31: K+8
// (K=0..7 and K=16..23 are each 16 contiguous bytes -> 2x ds_load_b128)
__device__ __forceinline__ v16h load_a_frag(const half_t* src, int ld, int k0) {
    int lane = threadIdx.x & 31;
    int m = lane & 15;
    int khalf = (lane >> 4) * 8;
    v16h a;
#pragma unroll
    for (int g = 0; g < 8; ++g) {
        int kbase = (g < 4) ? (2 * g) : (16 + 2 * (g - 4));
        int k = k0 + khalf + kbase;
        a[2 * g]     = src[m * ld + k];
        a[2 * g + 1] = src[m * ld + k + 1];
    }
    return a;
}

// B fragment from LDS-staged pre-swizzled pool: one contiguous 32B read per lane
__device__ __forceinline__ v16h load_b_frag(const half_t* frags, int fragIdx) {
    int lane = threadIdx.x & 31;
    return *(const v16h*)(frags + ((size_t)fragIdx * 32 + lane) * 16);
}

// ---------------------------------------------------------------- prep kernel
// f32 row-major W[K][N] -> f16 WMMA B fragments (same K order as A frags:
// lane = N column, VGPR0-3: K=0..7, VGPR4-7: K=16..23; lanes 16-31: K+8)
__device__ __forceinline__ void swizzle_b_frag(const float* W, int N, int kt, int nt,
                                               int lane, half_t* dst) {
    int n = nt * 16 + (lane & 15);
    int khalf = (lane >= 16) ? 8 : 0;
#pragma unroll
    for (int g = 0; g < 8; ++g) {
        int k = kt * 32 + khalf + ((g < 4) ? (2 * g) : (16 + 2 * (g - 4)));
        dst[2 * g]     = (half_t)W[k * N + n];
        dst[2 * g + 1] = (half_t)W[(k + 1) * N + n];
    }
}

__global__ void prep_kernel(const float* Wt0, const float* Wt1, const float* Wt2,
                            const float* Wb1, half_t* swAll) {
    int t = blockIdx.x * blockDim.x + threadIdx.x;
    int lane = t & 31;
    int frag = t >> 5;
    if (frag < 32) {                       // Wt0: 64x256, Kt=2, Nt=16, fragIdx=nt*2+kt
        int nt = frag >> 1, kt = frag & 1;
        swizzle_b_frag(Wt0, 256, kt, nt, lane,
                       swAll + OFF_W0 + ((size_t)frag * 32 + lane) * 16);
    } else if (frag < 96) {                // Wt1: 256x128, Kt=8, Nt=8, fragIdx=nt*8+kt
        int f = frag - 32;
        int nt = f >> 3, kt = f & 7;
        swizzle_b_frag(Wt1, 128, kt, nt, lane,
                       swAll + OFF_W1 + ((size_t)f * 32 + lane) * 16);
    } else if (frag < 112) {               // Wt2: 128x64, Kt=4, Nt=4, fragIdx=nt*4+kt
        int f = frag - 96;
        int nt = f >> 2, kt = f & 3;
        swizzle_b_frag(Wt2, 64, kt, nt, lane,
                       swAll + OFF_W2 + ((size_t)f * 32 + lane) * 16);
    } else if (frag < 114) {               // Wb1: 32x32, Kt=1, Nt=2
        int f = frag - 112;
        swizzle_b_frag(Wb1, 32, 0, f, lane,
                       swAll + OFF_WB1 + ((size_t)f * 32 + lane) * 16);
    }
}

// ---------------------------------------------------------------- user vector
__global__ void user_kernel(const int* __restrict__ hist, const int* __restrict__ wish,
                            const float* __restrict__ Eh, const float* __restrict__ Ew,
                            half_t* __restrict__ uf16) {
    int b = blockIdx.x * 8 + (threadIdx.x >> 5);   // wave-uniform
    int d = threadIdx.x & 31;
    const int* hp = hist + b * 200;
    const int* wp = wish + b * 50;
    float s1 = 0.f, s2 = 0.f;
    for (int l = 0; l < 200; ++l) s1 += Eh[hp[l] * 32 + d];
    for (int l = 0; l < 50;  ++l) s2 += Ew[wp[l] * 32 + d];
    uf16[b * 32 + d] = (half_t)(s1 * (1.f / 200.f) + s2 * (1.f / 50.f));
}

// ---------------------------------------------------------------- fused item + top MLP
// Persistent blocks: 8 waves/block, each wave owns an LDS slab and grid-strides
// over 32-row super-tiles. All weights staged once per block into LDS.
__global__ __launch_bounds__(256) void dlrm_top_kernel(
    const int* __restrict__ cand, const int* __restrict__ auth,
    const int* __restrict__ lang, const int* __restrict__ tags,
    const float* __restrict__ dense,
    const float* __restrict__ Ec, const float* __restrict__ Ea,
    const float* __restrict__ El, const float* __restrict__ Et,
    const float* __restrict__ Wb0, const float* __restrict__ bb0,
    const float* __restrict__ bb1,
    const float* __restrict__ bt0, const float* __restrict__ bt1,
    const float* __restrict__ bt2, const float* __restrict__ Wt3,
    const float* __restrict__ bt3,
    const half_t* __restrict__ uf16, const half_t* __restrict__ swAll,
    float* __restrict__ out) {

    __shared__ __align__(32) half_t wlds[W_HALVES];
    __shared__ __align__(32) half_t slabs[TOP_WAVES][SLAB_HALVES];

    int lane = threadIdx.x & 31;
    int waveId = threadIdx.x >> 5;

    // stage all swizzled weights into LDS (116,736 B, once per block)
    {
        const uint4* s = (const uint4*)swAll;
        uint4* d = (uint4*)wlds;
        for (int i = threadIdx.x; i < W_HALVES / 8; i += blockDim.x) d[i] = s[i];
    }
    __syncthreads();

    const half_t* wW0  = wlds + OFF_W0;
    const half_t* wW1  = wlds + OFF_W1;
    const half_t* wW2  = wlds + OFF_W2;
    const half_t* wWb1 = wlds + OFF_WB1;

    half_t* slab = &slabs[waveId][0];
    half_t* xh   = slab;             // [32][64], later reused as h3
    half_t* h1   = slab + 2048;      // [32][256]
    half_t* hbuf = h1;               // [32][32]  (dead once stage 2 reads it)
    half_t* h2   = h1;               // [32][128] (h1 dead after L1 A-frag loads)
    half_t* h3   = xh;               // [32][64]  (xh dead after L0 A-frag loads)
    float*  ivec = (float*)(slab + 10240); // [32][32] f32

    int col = lane & 15;
    int rhalf = (lane >> 4) * 8;

    float wb00 = Wb0[lane], wb01 = Wb0[32 + lane], wb02 = Wb0[64 + lane];
    float bb0j = bb0[lane];
    float bb1j = bb1[lane];

    for (int st = blockIdx.x * TOP_WAVES + waveId; st < NSUPER;
         st += gridDim.x * TOP_WAVES) {
        int base = st * 32;

        // ---- stage 1: gathers, user cols, bottom layer-0 ----
        for (int r = 0; r < 32; ++r) {
            int row = base + r;                 // wave-uniform
            int b = row / C_DIM;
            int ic = cand[row], ia = auth[row], il = lang[row];
            float v = Ec[ic * 32 + lane] + Ea[ia * 32 + lane] + El[il * 32 + lane];
            const int* tg = tags + row * 5;
            float ts = 0.f;
#pragma unroll
            for (int t = 0; t < 5; ++t) ts += Et[tg[t] * 32 + lane];
            ivec[r * 32 + lane] = v + 0.2f * ts + bb1j;           // i_vec + bb1
            xh[r * 64 + lane] = uf16[b * 32 + lane];              // cols 0..31
            float d0 = dense[row * 3 + 0], d1 = dense[row * 3 + 1],
                  d2 = dense[row * 3 + 2];
            float h = fmaf(d0, wb00, fmaf(d1, wb01, fmaf(d2, wb02, bb0j)));
            hbuf[r * 32 + lane] = (half_t)fmaxf(h, 0.f);
        }

        // ---- stage 2: d_vec = H @ Wb1 (K=32); xh cols 32..63 = ivec + d_vec ----
        {
            v16h a0 = load_a_frag(hbuf, 32, 0);
            v16h a1 = load_a_frag(hbuf + 16 * 32, 32, 0);
#pragma unroll
            for (int nt = 0; nt < 2; ++nt) {
                v16h bf = load_b_frag(wWb1, nt);
                v8f c0 = {}, c1 = {};
                c0 = wmma16(a0, bf, c0);
                c1 = wmma16(a1, bf, c1);
                int n = nt * 16 + col;
#pragma unroll
                for (int g = 0; g < 8; ++g) {
                    int rr = g + rhalf;
                    xh[rr * 64 + 32 + n] = (half_t)(c0[g] + ivec[rr * 32 + n]);
                    xh[(16 + rr) * 64 + 32 + n] =
                        (half_t)(c1[g] + ivec[(16 + rr) * 32 + n]);
                }
            }
        }

        // ---- layer 0: [32,64] @ [64,256] -> h1 ----
        {
            v16h a00 = load_a_frag(xh, 64, 0);
            v16h a01 = load_a_frag(xh, 64, 32);
            v16h a10 = load_a_frag(xh + 16 * 64, 64, 0);
            v16h a11 = load_a_frag(xh + 16 * 64, 64, 32);
#pragma unroll
            for (int nt = 0; nt < 16; ++nt) {
                float bias = bt0[nt * 16 + col];
                v8f c0, c1;
#pragma unroll
                for (int g = 0; g < 8; ++g) { c0[g] = bias; c1[g] = bias; }
                v16h b0 = load_b_frag(wW0, nt * 2 + 0);
                v16h b1 = load_b_frag(wW0, nt * 2 + 1);
                c0 = wmma16(a00, b0, c0);
                c0 = wmma16(a01, b1, c0);
                c1 = wmma16(a10, b0, c1);
                c1 = wmma16(a11, b1, c1);
#pragma unroll
                for (int g = 0; g < 8; ++g) {
                    h1[(g + rhalf) * 256 + nt * 16 + col] = (half_t)fmaxf(c0[g], 0.f);
                    h1[(16 + g + rhalf) * 256 + nt * 16 + col] =
                        (half_t)fmaxf(c1[g], 0.f);
                }
            }
        }

        // ---- layer 1: [32,256] @ [256,128] -> h2 (aliases h1; h1 is fully
        //      consumed into registers before the first h2 store) ----
        {
            v16h a[2][8];
#pragma unroll
            for (int kt = 0; kt < 8; ++kt) {
                a[0][kt] = load_a_frag(h1, 256, kt * 32);
                a[1][kt] = load_a_frag(h1 + 16 * 256, 256, kt * 32);
            }
#pragma unroll
            for (int nt = 0; nt < 8; ++nt) {
                float bias = bt1[nt * 16 + col];
                v8f c0, c1;
#pragma unroll
                for (int g = 0; g < 8; ++g) { c0[g] = bias; c1[g] = bias; }
#pragma unroll
                for (int kt = 0; kt < 8; ++kt) {
                    v16h bf = load_b_frag(wW1, nt * 8 + kt);
                    c0 = wmma16(a[0][kt], bf, c0);
                    c1 = wmma16(a[1][kt], bf, c1);
                }
#pragma unroll
                for (int g = 0; g < 8; ++g) {
                    h2[(g + rhalf) * 128 + nt * 16 + col] = (half_t)fmaxf(c0[g], 0.f);
                    h2[(16 + g + rhalf) * 128 + nt * 16 + col] =
                        (half_t)fmaxf(c1[g], 0.f);
                }
            }
        }

        // ---- layer 2: [32,128] @ [128,64] -> h3 (aliases xh) ----
        {
            v16h a[2][4];
#pragma unroll
            for (int kt = 0; kt < 4; ++kt) {
                a[0][kt] = load_a_frag(h2, 128, kt * 32);
                a[1][kt] = load_a_frag(h2 + 16 * 128, 128, kt * 32);
            }
#pragma unroll
            for (int nt = 0; nt < 4; ++nt) {
                float bias = bt2[nt * 16 + col];
                v8f c0, c1;
#pragma unroll
                for (int g = 0; g < 8; ++g) { c0[g] = bias; c1[g] = bias; }
#pragma unroll
                for (int kt = 0; kt < 4; ++kt) {
                    v16h bf = load_b_frag(wW2, nt * 4 + kt);
                    c0 = wmma16(a[0][kt], bf, c0);
                    c1 = wmma16(a[1][kt], bf, c1);
                }
#pragma unroll
                for (int g = 0; g < 8; ++g) {
                    h3[(g + rhalf) * 64 + nt * 16 + col] = (half_t)fmaxf(c0[g], 0.f);
                    h3[(16 + g + rhalf) * 64 + nt * 16 + col] =
                        (half_t)fmaxf(c1[g], 0.f);
                }
            }
        }

        // ---- layer 3: [32,64] @ [64,1] via lane-pair dot + shfl reduce ----
#pragma unroll
        for (int hb = 0; hb < 2; ++hb) {
            int r = hb * 16 + (lane >> 1);
            int hh = lane & 1;
            float s = 0.f;
#pragma unroll
            for (int k = 0; k < 32; ++k)
                s += (float)h3[r * 64 + hh * 32 + k] * Wt3[hh * 32 + k];
            s += __shfl_xor(s, 1, 32);
            if (hh == 0) out[base + r] = s + bt3[0];
        }
    }
}

// ---------------------------------------------------------------- launch

extern "C" void kernel_launch(void* const* d_in, const int* in_sizes, int n_in,
                              void* d_out, int out_size, void* d_ws, size_t ws_size,
                              hipStream_t stream) {
    const int*   hist  = (const int*)d_in[0];
    const int*   wish  = (const int*)d_in[1];
    const int*   cand  = (const int*)d_in[2];
    const int*   auth  = (const int*)d_in[3];
    const int*   lang  = (const int*)d_in[4];
    const int*   tags  = (const int*)d_in[5];
    const float* dense = (const float*)d_in[6];
    const float* Eh    = (const float*)d_in[7];
    const float* Ew    = (const float*)d_in[8];
    const float* Ec    = (const float*)d_in[9];
    const float* Ea    = (const float*)d_in[10];
    const float* El    = (const float*)d_in[11];
    const float* Et    = (const float*)d_in[12];
    const float* Wb0   = (const float*)d_in[13];
    const float* bb0   = (const float*)d_in[14];
    const float* Wb1   = (const float*)d_in[15];
    const float* bb1   = (const float*)d_in[16];
    const float* Wt0   = (const float*)d_in[17];
    const float* bt0   = (const float*)d_in[18];
    const float* Wt1   = (const float*)d_in[19];
    const float* bt1   = (const float*)d_in[20];
    const float* Wt2   = (const float*)d_in[21];
    const float* bt2   = (const float*)d_in[22];
    const float* Wt3   = (const float*)d_in[23];
    const float* bt3   = (const float*)d_in[24];

    // workspace: [swAll 58368 halves | uf16 131072 halves]
    half_t* swAll = (half_t*)d_ws;
    half_t* uf16  = swAll + W_HALVES;

    prep_kernel<<<15, 256, 0, stream>>>(Wt0, Wt1, Wt2, Wb1, swAll);
    user_kernel<<<B_DIM / 8, 256, 0, stream>>>(hist, wish, Eh, Ew, uf16);
    dlrm_top_kernel<<<TOP_BLOCKS, 256, 0, stream>>>(
        cand, auth, lang, tags, dense, Ec, Ea, El, Et,
        Wb0, bb0, bb1, bt0, bt1, bt2, Wt3, bt3,
        uf16, swAll, (float*)d_out);
}
